// MemoryAugmentedAttention_69827578298917
// MI455X (gfx1250) — matrix operand order repaired and checked by reading, not compiled, not run on hardware
//
#include <hip/hip_runtime.h>
#include <hip/hip_bf16.h>
#include <stdint.h>

#define HEADS 16
#define HD 64
#define TLEN 2048
#define BATCH 2
#define DMODEL 1024
#define MEMN 512
#define KTOT (MEMN + TLEN) /* 2560 */

typedef __attribute__((ext_vector_type(16))) __bf16 v16bf;
typedef __attribute__((ext_vector_type(8)))  float  v8f;
typedef __attribute__((ext_vector_type(4)))  int    v4i;

#define AS1 __attribute__((address_space(1)))
#define AS3 __attribute__((address_space(3)))

union AFrag { v16bf v; uint4 u[2]; };

__device__ inline unsigned short f2bf(float f) {
  unsigned int u = __float_as_uint(f);
  u += 0x7FFFu + ((u >> 16) & 1u);
  return (unsigned short)(u >> 16);
}
__device__ inline float bf2f(unsigned short h) {
  return __uint_as_float(((unsigned int)h) << 16);
}

// 16-byte async copy global -> LDS (gfx1250 GLOBAL_LOAD_ASYNC_TO_LDS_B128).
// Generic LDS pointers keep the DS offset in addr[31:0] (flat aperture rules),
// so the addrspace(3) pointer is recovered by truncation.
__device__ inline void async_ld16(const unsigned short* g, unsigned short* l) {
#if __has_builtin(__builtin_amdgcn_global_load_async_to_lds_b128)
  __builtin_amdgcn_global_load_async_to_lds_b128(
      (AS1 v4i*)(uintptr_t)g, (AS3 v4i*)(unsigned int)(uintptr_t)l, 0, 0);
#else
  *(uint4*)l = *(const uint4*)g;
#endif
}
__device__ inline void wait_async() {
#if __has_builtin(__builtin_amdgcn_s_wait_asynccnt)
  __builtin_amdgcn_s_wait_asynccnt(0);
#else
  asm volatile("s_wait_asynccnt 0" ::: "memory");
#endif
}

// ---------------------------------------------------------------- cvt fp32->bf16
__global__ void k_cvt_bf16(const float* __restrict__ src,
                           unsigned short* __restrict__ dst, int n) {
  int i = blockIdx.x * blockDim.x + threadIdx.x;
  if (i < n) dst[i] = f2bf(src[i]);
}

// ---------------------------------------------------------------- bf16 GEMM
// C[M,N] = A[M,K] * B[N,K]^T ; A,B row-major bf16. Each wave: 64x64 tile
// (4x4 WMMA accumulators; B fragments held in registers, A streamed).
template <int WRITE_BF16>
__global__ void __launch_bounds__(128)
k_gemm_bf16(const unsigned short* __restrict__ A,
            const unsigned short* __restrict__ Bm,
            unsigned short* __restrict__ Cb,
            float* __restrict__ Cf,
            const float* __restrict__ bias,
            int M, int N, int K) {
  const int lane = threadIdx.x & 31;
  const int wave = threadIdx.x >> 5;
  const int ln = lane & 15, lh = lane >> 4;
  const int wm = wave >> 1, wn = wave & 1;
  const int m0 = blockIdx.y * 128 + wm * 64;
  const int n0 = blockIdx.x * 128 + wn * 64;
  const int alo = lh * 8;

  v8f acc[4][4] = {};
  for (int k = 0; k < K; k += 32) {
    AFrag b[4];
#pragma unroll
    for (int ni = 0; ni < 4; ++ni) {
      const unsigned short* bp = Bm + (size_t)(n0 + ni * 16 + ln) * K + k;
      b[ni].u[0] = *(const uint4*)(bp + alo);
      b[ni].u[1] = *(const uint4*)(bp + 16 + alo);
    }
#pragma unroll
    for (int mi = 0; mi < 4; ++mi) {
      AFrag a;
      const unsigned short* ap = A + (size_t)(m0 + mi * 16 + ln) * K + k;
      a.u[0] = *(const uint4*)(ap + alo);
      a.u[1] = *(const uint4*)(ap + 16 + alo);
#pragma unroll
      for (int ni = 0; ni < 4; ++ni)
        acc[mi][ni] = __builtin_amdgcn_wmma_f32_16x16x32_bf16(
            false, a.v, false, b[ni].v, (short)0, acc[mi][ni], false, false);
    }
  }
#pragma unroll
  for (int mi = 0; mi < 4; ++mi)
#pragma unroll
    for (int ni = 0; ni < 4; ++ni)
#pragma unroll
      for (int i = 0; i < 8; ++i) {
        int r = m0 + mi * 16 + i + 8 * lh;
        int c = n0 + ni * 16 + ln;
        if (WRITE_BF16)
          Cb[(size_t)r * N + c] = f2bf(acc[mi][ni][i]);
        else
          Cf[(size_t)r * N + c] = acc[mi][ni][i] + bias[c];
      }
}

// ---------------------------------------------------------------- split QKV
__global__ void k_split_qkv(const unsigned short* __restrict__ qkv,
                            unsigned short* __restrict__ Qb,
                            unsigned short* __restrict__ Kb,
                            unsigned short* __restrict__ Vt) {
  int idx = blockIdx.x * blockDim.x + threadIdx.x;
  const int total = BATCH * TLEN * 3 * DMODEL;
  if (idx >= total) return;
  int e = idx % (3 * DMODEL);
  int bt = idx / (3 * DMODEL);
  int b = bt / TLEN, t = bt % TLEN;
  int part = e / DMODEL;
  int c = e % DMODEL;
  int h = c / HD, d = c % HD;
  unsigned short v = qkv[idx];
  size_t bh = (size_t)b * HEADS + h;
  if (part == 0) {
    Qb[(bh * TLEN + t) * HD + d] = f2bf(bf2f(v) * 0.125f); // hd^-0.5
  } else if (part == 1) {
    Kb[(bh * KTOT + MEMN + t) * HD + d] = v;
  } else {
    Vt[(bh * HD + d) * KTOT + MEMN + t] = v;
  }
}

// ---------------------------------------------------------------- memory K/V fill
__global__ void k_mem_kv(const float* __restrict__ memory,
                         unsigned short* __restrict__ Kb,
                         unsigned short* __restrict__ Vt) {
  int idx = blockIdx.x * blockDim.x + threadIdx.x;
  const int total = BATCH * HEADS * MEMN * HD;
  if (idx >= total) return;
  int d = idx % HD;
  int m = (idx / HD) % MEMN;
  int h = (idx / (HD * MEMN)) % HEADS;
  int b = idx / (HD * MEMN * HEADS);
  unsigned short v = f2bf(memory[(size_t)m * DMODEL + h * HD + d]);
  size_t bh = (size_t)b * HEADS + h;
  Kb[(bh * KTOT + m) * HD + d] = v;
  Vt[(bh * HD + d) * KTOT + m] = v;
}

// ---------------------------------------------------------------- flash attention
// 128 threads = 4 waves; each wave owns 16 query rows. K/V 64x64 tiles are
// staged into LDS once per key-block via async load-to-LDS and shared by all
// 4 waves (4x traffic reduction vs. per-wave global fragment loads).
__global__ void __launch_bounds__(128)
k_flash_attn(const unsigned short* __restrict__ Qb,
             const unsigned short* __restrict__ Kb,
             const unsigned short* __restrict__ Vt,
             unsigned short* __restrict__ Ob) {
  __shared__ unsigned short ldsK[64][64];   // [key][hd]
  __shared__ unsigned short ldsV[64][64];   // [hd][key-chunk]
  __shared__ unsigned short ldsP[4][16][64];
  const int tid = threadIdx.x;
  const int lane = tid & 31;
  const int wave = tid >> 5;
  const int ln = lane & 15, lh = lane >> 4;
  const int alo = lh * 8;

  const int qblk = blockIdx.x % (TLEN / 64);
  const int bh = blockIdx.x / (TLEN / 64);
  const int b = bh / HEADS;
  const int h = bh % HEADS;
  const int q0 = qblk * 64 + wave * 16;

  const unsigned short* Qp = Qb + ((size_t)bh * TLEN + q0 + ln) * HD;
  const unsigned short* Kbase = Kb + (size_t)bh * KTOT * HD;
  const unsigned short* Vbase = Vt + (size_t)bh * HD * KTOT;

  AFrag aq[2];
#pragma unroll
  for (int kc = 0; kc < 2; ++kc) {
    aq[kc].u[0] = *(const uint4*)(Qp + kc * 32 + alo);
    aq[kc].u[1] = *(const uint4*)(Qp + kc * 32 + 16 + alo);
  }

  float mrow[8], lrow[8];
  v8f acc[4] = {};
#pragma unroll
  for (int i = 0; i < 8; ++i) { mrow[i] = -1e30f; lrow[i] = 0.f; }

  for (int kb = 0; kb < KTOT / 64; ++kb) {
    // ---- stage K and V tiles (8KB each) into LDS, async
#pragma unroll
    for (int i = 0; i < 4; ++i) {
      int c = tid + i * 128;       // 0..511 16B-chunks
      int row = c >> 3;            // 0..63
      int col = (c & 7) * 8;       // 0..56 (elements)
      async_ld16(Kbase + (size_t)(kb * 64 + row) * HD + col, &ldsK[row][col]);
      async_ld16(Vbase + (size_t)row * KTOT + kb * 64 + col, &ldsV[row][col]);
    }
    wait_async();
    __syncthreads();

    // ---- S = Q K^T (16 x 64 per wave), fragments from LDS
    v8f s[4];
#pragma unroll
    for (int nt = 0; nt < 4; ++nt) {
      v8f c = {};
#pragma unroll
      for (int kc = 0; kc < 2; ++kc) {
        AFrag bk;
        const unsigned short* kp = &ldsK[nt * 16 + ln][kc * 32];
        bk.u[0] = *(const uint4*)(kp + alo);
        bk.u[1] = *(const uint4*)(kp + 16 + alo);
        c = __builtin_amdgcn_wmma_f32_16x16x32_bf16(false, aq[kc].v, false, bk.v,
                                                    (short)0, c, false, false);
      }
      s[nt] = c;
    }
    // ---- online softmax (row lives in one acc VGPR across a 16-lane half)
    float p[4][8];
#pragma unroll
    for (int i = 0; i < 8; ++i) {
      float v = fmaxf(fmaxf(s[0][i], s[1][i]), fmaxf(s[2][i], s[3][i]));
#pragma unroll
      for (int off = 1; off < 16; off <<= 1) v = fmaxf(v, __shfl_xor(v, off, 32));
      float mn = fmaxf(mrow[i], v);
      float alpha = __expf(mrow[i] - mn);
      mrow[i] = mn;
      float rs = 0.f;
#pragma unroll
      for (int nt = 0; nt < 4; ++nt) {
        float e = __expf(s[nt][i] - mn);
        p[nt][i] = e;
        rs += e;
      }
#pragma unroll
      for (int off = 1; off < 16; off <<= 1) rs += __shfl_xor(rs, off, 32);
      lrow[i] = lrow[i] * alpha + rs;
#pragma unroll
      for (int nt = 0; nt < 4; ++nt) acc[nt][i] *= alpha;
    }
    // ---- bounce P through LDS: C-layout -> A-operand layout
#pragma unroll
    for (int nt = 0; nt < 4; ++nt)
#pragma unroll
      for (int i = 0; i < 8; ++i)
        ldsP[wave][i + 8 * lh][nt * 16 + ln] = f2bf(p[nt][i]);
    __syncthreads();
    // ---- O += P V, V fragments from LDS
#pragma unroll
    for (int kc = 0; kc < 2; ++kc) {
      AFrag ap;
      const unsigned short* pp = &ldsP[wave][ln][kc * 32];
      ap.u[0] = *(const uint4*)(pp + alo);
      ap.u[1] = *(const uint4*)(pp + 16 + alo);
#pragma unroll
      for (int nt = 0; nt < 4; ++nt) {
        AFrag bv;
        const unsigned short* vp = &ldsV[nt * 16 + ln][kc * 32];
        bv.u[0] = *(const uint4*)(vp + alo);
        bv.u[1] = *(const uint4*)(vp + 16 + alo);
        acc[nt] = __builtin_amdgcn_wmma_f32_16x16x32_bf16(
            false, ap.v, false, bv.v, (short)0, acc[nt], false, false);
      }
    }
    __syncthreads();  // protect ldsK/ldsV/ldsP before next stage
  }
  // ---- write attention output [B,T,D] bf16
#pragma unroll
  for (int nt = 0; nt < 4; ++nt)
#pragma unroll
    for (int i = 0; i < 8; ++i) {
      int row = i + 8 * lh;
      int col = h * HD + nt * 16 + ln;
      float o = acc[nt][i] / lrow[i];
      Ob[((size_t)b * TLEN + q0 + row) * DMODEL + col] = f2bf(o);
    }
}

// ---------------------------------------------------------------- launch
extern "C" void kernel_launch(void* const* d_in, const int* in_sizes, int n_in,
                              void* d_out, int out_size, void* d_ws, size_t ws_size,
                              hipStream_t stream) {
  const float* x      = (const float*)d_in[0];
  const float* w_qkv  = (const float*)d_in[1];
  const float* w_out  = (const float*)d_in[2];
  const float* b_out  = (const float*)d_in[3];
  const float* memory = (const float*)d_in[4];
  float* out = (float*)d_out;

  char* ws = (char*)d_ws;
  unsigned short* xb    = (unsigned short*)(ws + 0);         //  8 MB
  unsigned short* wqkvb = (unsigned short*)(ws + 8388608);   //  6 MB
  unsigned short* woutb = (unsigned short*)(ws + 14680064);  //  2 MB
  unsigned short* qkvb  = (unsigned short*)(ws + 16777216);  // 24 MB
  unsigned short* Qb    = (unsigned short*)(ws + 41943040);  //  8 MB
  unsigned short* Kbuf  = (unsigned short*)(ws + 50331648);  // 10 MB
  unsigned short* Vt    = (unsigned short*)(ws + 60817408);  // 10 MB
  unsigned short* attnb = (unsigned short*)(ws + 71303168);  //  8 MB

  const int nx = BATCH * TLEN * DMODEL;
  const int nqw = 3 * DMODEL * DMODEL;
  const int now = DMODEL * DMODEL;
  k_cvt_bf16<<<(nx + 255) / 256, 256, 0, stream>>>(x, xb, nx);
  k_cvt_bf16<<<(nqw + 255) / 256, 256, 0, stream>>>(w_qkv, wqkvb, nqw);
  k_cvt_bf16<<<(now + 255) / 256, 256, 0, stream>>>(w_out, woutb, now);

  // QKV projection: M=4096, N=3072, K=1024 -> bf16
  dim3 g1(3072 / 128, 4096 / 128);
  k_gemm_bf16<1><<<g1, 128, 0, stream>>>(xb, wqkvb, qkvb, nullptr, nullptr,
                                         BATCH * TLEN, 3 * DMODEL, DMODEL);

  const int nsplit = BATCH * TLEN * 3 * DMODEL;
  k_split_qkv<<<(nsplit + 255) / 256, 256, 0, stream>>>(qkvb, Qb, Kbuf, Vt);

  const int nmem = BATCH * HEADS * MEMN * HD;
  k_mem_kv<<<(nmem + 255) / 256, 256, 0, stream>>>(memory, Kbuf, Vt);

  k_flash_attn<<<BATCH * HEADS * (TLEN / 64), 128, 0, stream>>>(Qb, Kbuf, Vt, attnb);

  // Output projection + bias: M=4096, N=1024, K=1024 -> fp32
  dim3 g2(1024 / 128, 4096 / 128);
  k_gemm_bf16<0><<<g2, 128, 0, stream>>>(attnb, woutb, nullptr, out, b_out,
                                         BATCH * TLEN, DMODEL, DMODEL);
}
